// DAtt_ZS_38835094291120
// MI455X (gfx1250) — matrix operand order repaired and checked by reading, not compiled, not run on hardware
//
#include <hip/hip_runtime.h>
#include <hip/hip_bf16.h>

// ---------------------------------------------------------------------------
// MI455X (gfx1250) implementation of DAtt_ZS forward.
//
// Roofline: all [B,64,T] activations are L2-resident (192MB L2); HBM traffic
// is just x (5.2MB) + out (0.5MB). Compute ~20 GFLOP after algebraic collapse:
//   scores = Wq G Wk^T + rank-1 bias terms, G = H H^T  (only T-contraction)
//   o1 = (S Wv1) h' + S bv1 = M1 h' + n1   (V never materialized)
//   final 3 linear layers -> single 64-dot  y = sigmoid(w_eff . taff + b_eff)
// WMMA (v_wmma_f32_16x16x32_f16, f32 accumulate) covers: TCN block 3 (5.3
// GFLOP, dilated convs as GEMMs with K = k*Cin+ci ordering so B-fragments are
// contiguous 32B LDS reads from a channel-transposed tile), the Gram matrix
// (1.07 GFLOP), and the fused o1/o2 + CAM pass (2.8 GFLOP). Only the small
// 10->16->32 TCN blocks (~2.3 GFLOP) stay VALU fp32.
// ---------------------------------------------------------------------------

typedef _Float16 v16h  __attribute__((ext_vector_type(16)));
typedef _Float16 f16x8 __attribute__((ext_vector_type(8)));
typedef float    v8f   __attribute__((ext_vector_type(8)));

constexpr int B_ = 16, T_ = 8192;
constexpr float BN_SCALE_F = 0.9999950000374998f; // 1/sqrt(1+1e-5)

__device__ __forceinline__ v8f wmma_f16(v16h a, v16h b, v8f c) {
  return __builtin_amdgcn_wmma_f32_16x16x32_f16(false, a, false, b,
                                                (short)0, c, false, false);
}
// 16-bit A-fragment: lane half hi picks local K {0-15} vs {16-31}
__device__ __forceinline__ v16h load_afrag(const _Float16* row, int off, int hi) {
  union { v16h v; f16x8 h[2]; } A;
  A.h[0] = *(const f16x8*)(row + off + hi * 16);
  A.h[1] = *(const f16x8*)(row + off + hi * 16 + 8);
  return A.v;
}

// ---- workspace layout (bytes) ----
constexpr size_t OFF_H16   = 0;                               // [B][64][T] f16
constexpr size_t SZ_H16    = (size_t)B_ * 64 * T_ * 2;        // 16 MB
constexpr size_t OFF_H16T  = OFF_H16 + SZ_H16;                // [B][T][64] f16
constexpr size_t OFF_GPART = OFF_H16T + SZ_H16;               // [8][B][64][64] f32
constexpr size_t SZ_GPART  = (size_t)8 * B_ * 64 * 64 * 4;
constexpr size_t OFF_HSUM  = OFF_GPART + SZ_GPART;            // [B][64] f32
constexpr size_t OFF_M1    = OFF_HSUM + (size_t)B_ * 64 * 4;  // [B][64][64] f16
constexpr size_t OFF_M2    = OFF_M1 + (size_t)B_ * 4096 * 2;
constexpr size_t OFF_N1    = OFF_M2 + (size_t)B_ * 4096 * 2;  // [B][64] f32
constexpr size_t OFF_N2    = OFF_N1 + (size_t)B_ * 64 * 4;
constexpr size_t OFF_AG    = OFF_N2 + (size_t)B_ * 64 * 4;    // gate a-branch
constexpr size_t OFF_C3F   = OFF_AG + (size_t)B_ * 64 * 4;    // [16][64] f16 (BN folded)
constexpr size_t OFF_C4F   = OFF_C3F + 16 * 64 * 2;           // [64][16] f16 (BN folded)
constexpr size_t OFF_B3    = OFF_C4F + 64 * 16 * 2;           // [16] f32
constexpr size_t OFF_B4    = OFF_B3 + 16 * 4;                 // [64] f32
constexpr size_t OFF_WEFF  = OFF_B4 + 64 * 4;                 // [64] f32
constexpr size_t OFF_BEFF  = OFF_WEFF + 64 * 4;               // [1] f32
constexpr size_t OFF_W1CP  = OFF_BEFF + 32;                   // [64][96]  f16, K=k*32+ci
constexpr size_t OFF_W2CP  = OFF_W1CP + 64 * 96 * 2;          // [64][192] f16, K=k*64+ci
constexpr size_t OFF_WDCP  = OFF_W2CP + 64 * 192 * 2;         // [64][32]  f16, K=ci

// ===========================================================================
// K1: fused causal TCN (3 blocks) + positional-encoding quirk.
// 128-t tile, 28-sample left halo. Blocks 1-2 (10->16->32) VALU fp32 in
// phased-reuse LDS; block 3 (32->64, d=4) runs on WMMA from channel-
// transposed f16 LDS tiles. Columns mapping to global t<0 forced to 0
// (reference zero-pads each conv input).
// ===========================================================================
__global__ void tcn_kernel(
    const float* __restrict__ x, const float* __restrict__ pe,
    const float* w1a, const float* b1a, const float* w2a, const float* b2a,
    const float* wda, const float* bda,
    const float* w1b, const float* b1b, const float* w2b, const float* b2b,
    const float* wdb, const float* bdb,
    const float* b1c, const float* b2c, const float* bdc,
    const _Float16* __restrict__ w1cp, const _Float16* __restrict__ w2cp,
    const _Float16* __restrict__ wdcp,
    _Float16* __restrict__ h16, _Float16* __restrict__ h16T)
{
  constexpr int TS = 128;
  constexpr int NX = TS + 28, NA1 = TS + 26, NH1 = TS + 24;
  constexpr int NA2 = TS + 20, NH2 = TS + 16, NA3 = TS + 8;
  __shared__ __align__(64) char smem[63232];
  float* xs = (float*)(smem + 0);       // [10][NX]
  float* a1 = (float*)(smem + 6240);    // [16][NA1]
  float* h1 = (float*)(smem + 16096);   // [16][NH1]
  float* a2 = (float*)(smem + 25824);   // [32][NA2]
  float* h2 = (float*)(smem + 44768);   // [32][NH2]   (fp32, alive to the end)
  // f16 tiles for the WMMA stage (reuse dead xs/a1/h1 region):
  _Float16* h2T = (_Float16*)(smem + 0);     // [144][32]  h2 transposed (9216 B)
  _Float16* a3T = (_Float16*)(smem + 9216);  // [144][64]  a3 transposed (18432 B)

  const int b   = blockIdx.x;
  const int t0  = blockIdx.y * TS;
  const int tid = threadIdx.x;

  // stage input: xs[ci][j] <- x[b, t0-28+j, ci]
  for (int idx = tid; idx < 10 * NX; idx += 256) {
    int ci = idx / NX, j = idx % NX;
    int tg = t0 - 28 + j;
    xs[idx] = (tg >= 0) ? x[((size_t)b * T_ + tg) * 10 + ci] : 0.f;
  }
  __syncthreads();

  // block 1 (d=1): a1 = relu(conv1(x))
  for (int idx = tid; idx < 16 * NA1; idx += 256) {
    int co = idx / NA1, j = idx % NA1;
    int tg = t0 - 26 + j;
    float acc = b1a[co];
    for (int ci = 0; ci < 10; ++ci) {
      const float* wr = w1a + (co * 10 + ci) * 3;
      const float* xr = xs + ci * NX + j;
      acc = fmaf(wr[0], xr[0], acc);
      acc = fmaf(wr[1], xr[1], acc);
      acc = fmaf(wr[2], xr[2], acc);
    }
    a1[idx] = (tg >= 0) ? fmaxf(acc, 0.f) : 0.f;
  }
  __syncthreads();

  // h1 = relu(relu(conv2(a1)) + wd@x + bd)
  for (int idx = tid; idx < 16 * NH1; idx += 256) {
    int co = idx / NH1, j = idx % NH1;
    int tg = t0 - 24 + j;
    float acc = b2a[co];
    for (int ci = 0; ci < 16; ++ci) {
      const float* wr = w2a + (co * 16 + ci) * 3;
      const float* ar = a1 + ci * NA1 + j;
      acc = fmaf(wr[0], ar[0], acc);
      acc = fmaf(wr[1], ar[1], acc);
      acc = fmaf(wr[2], ar[2], acc);
    }
    float r = bda[co];
    for (int ci = 0; ci < 10; ++ci)
      r = fmaf(wda[co * 10 + ci], xs[ci * NX + j + 4], r);
    h1[idx] = (tg >= 0) ? fmaxf(fmaxf(acc, 0.f) + r, 0.f) : 0.f;
  }
  __syncthreads();

  // block 2 (d=2)
  for (int idx = tid; idx < 32 * NA2; idx += 256) {
    int co = idx / NA2, j = idx % NA2;
    int tg = t0 - 20 + j;
    float acc = b1b[co];
    for (int ci = 0; ci < 16; ++ci) {
      const float* wr = w1b + (co * 16 + ci) * 3;
      const float* hr = h1 + ci * NH1 + j;
      acc = fmaf(wr[0], hr[0], acc);
      acc = fmaf(wr[1], hr[2], acc);
      acc = fmaf(wr[2], hr[4], acc);
    }
    a2[idx] = (tg >= 0) ? fmaxf(acc, 0.f) : 0.f;
  }
  __syncthreads();

  for (int idx = tid; idx < 32 * NH2; idx += 256) {
    int co = idx / NH2, j = idx % NH2;
    int tg = t0 - 16 + j;
    float acc = b2b[co];
    for (int ci = 0; ci < 32; ++ci) {
      const float* wr = w2b + (co * 32 + ci) * 3;
      const float* ar = a2 + ci * NA2 + j;
      acc = fmaf(wr[0], ar[0], acc);
      acc = fmaf(wr[1], ar[2], acc);
      acc = fmaf(wr[2], ar[4], acc);
    }
    float r = bdb[co];
    for (int ci = 0; ci < 16; ++ci)
      r = fmaf(wdb[co * 16 + ci], h1[ci * NH1 + j + 8], r);
    h2[idx] = (tg >= 0) ? fmaxf(fmaxf(acc, 0.f) + r, 0.f) : 0.f;
  }
  __syncthreads();

  // channel-transposed f16 copy of h2 for WMMA B-fragments: h2T[col][ci]
  for (int idx = tid; idx < NH2 * 32; idx += 256) {
    int col = idx >> 5, ci = idx & 31;
    h2T[idx] = (_Float16)h2[ci * NH2 + col];
  }
  __syncthreads();

  const int lane = tid & 31, wave = tid >> 5;
  const int hi = (lane >> 4) & 1, ln = lane & 15;

  // block 3 conv1 (d=4, K=96 = k*32+ci): a3 = relu(W1c h2 + b1c), transposed
  // store into a3T. 8 waves x 16-col strips; wave 0 also covers cols 128-135
  // (reads past h2T's 144 cols stay inside smem and only feed unused cols).
  for (int jb = wave * 16; jb < NA3; jb += 128) {
#pragma unroll
    for (int cot = 0; cot < 4; ++cot) {
      const _Float16* arow = w1cp + (16 * cot + ln) * 96;
      v8f acc = {};
#pragma unroll
      for (int s = 0; s < 3; ++s) {   // K-step s <-> tap k=s (cols +4k)
        v16h A  = load_afrag(arow, s * 32, hi);
        v16h Bv = *(const v16h*)(h2T + (jb + ln + 4 * s) * 32 + hi * 16);
        acc = wmma_f16(A, Bv, acc);
      }
#pragma unroll
      for (int r = 0; r < 8; ++r) {   // C/D layout: row = r + 8*hi, col = ln
        int co = 16 * cot + r + 8 * hi;
        int j  = jb + ln;
        int tg = t0 - 8 + j;
        float v = fmaxf(acc[r] + b1c[co], 0.f);
        a3T[j * 64 + co] = (_Float16)((tg >= 0) ? v : 0.f);
      }
    }
  }
  __syncthreads();

  // block 3 conv2 (d=4, K=192 = k*64+ci) + residual (K=32) + epilogue
  {
    const int jb2 = wave * 16;
    v8f acc2[4] = {{}, {}, {}, {}};
    v8f racc[4] = {{}, {}, {}, {}};
#pragma unroll
    for (int cot = 0; cot < 4; ++cot) {
      const _Float16* arow = w2cp + (16 * cot + ln) * 192;
#pragma unroll
      for (int s = 0; s < 6; ++s) {
        int k = s >> 1, cib = (s & 1) * 32;
        v16h A  = load_afrag(arow, s * 32, hi);
        v16h Bv = *(const v16h*)(a3T + (jb2 + ln + 4 * k) * 64 + cib + hi * 16);
        acc2[cot] = wmma_f16(A, Bv, acc2[cot]);
      }
      const _Float16* rrow = wdcp + (16 * cot + ln) * 32;
      v16h Ar = load_afrag(rrow, 0, hi);
      v16h Br = *(const v16h*)(h2T + (jb2 + ln + 16) * 32 + hi * 16);
      racc[cot] = wmma_f16(Ar, Br, racc[cot]);
    }
#pragma unroll
    for (int cot = 0; cot < 4; ++cot)
#pragma unroll
      for (int r = 0; r < 8; ++r) {
        int co = 16 * cot + r + 8 * hi;
        int t  = t0 + jb2 + ln;
        float y  = fmaxf(acc2[cot][r] + b2c[co], 0.f);
        float hv = fmaxf(y + racc[cot][r] + bdc[co], 0.f) + pe[b * 64 + co];
        h16 [((size_t)b * 64 + co) * T_ + t] = (_Float16)hv;
        h16T[((size_t)b * T_ + t) * 64 + co] = (_Float16)hv;
      }
  }
}

// ===========================================================================
// K2: per-batch Gram matrix G = H H^T via WMMA f16->f32 (T split 8 ways into
// deterministic partial sums; no fp atomics so replays are bit-identical).
// 8 waves/WG each own 2 of the 16 (c,d) 16x16 tiles.
// ===========================================================================
__global__ void gram_kernel(const _Float16* __restrict__ h16,
                            float* __restrict__ gpart)
{
  const int b = blockIdx.x, part = blockIdx.y;
  const int tid = threadIdx.x, lane = tid & 31, wave = tid >> 5;
  const int ci = wave & 3, di0 = (wave >> 2) * 2;
  const int hi = (lane >> 4) & 1, ln = lane & 15;

  const _Float16* Hb    = h16 + (size_t)b * 64 * T_;
  const _Float16* Arow  = Hb + (size_t)(16 * ci + ln) * T_;
  const _Float16* Brow0 = Hb + (size_t)(16 * di0 + ln) * T_;
  const _Float16* Brow1 = Hb + (size_t)(16 * di0 + 16 + ln) * T_;

  v8f acc0 = {}, acc1 = {};
  const int tstart = part * (T_ / 8), tend = tstart + T_ / 8;
  for (int kt = tstart; kt < tend; kt += 32) {
    v16h A = load_afrag(Arow, kt, hi);
    // B in contiguous k order like A: lanes 0-15 K=0-15, lanes 16-31 K=16-31 (contig 32B)
    v16h Bf0 = *(const v16h*)(Brow0 + kt + hi * 16);
    v16h Bf1 = *(const v16h*)(Brow1 + kt + hi * 16);
    acc0 = wmma_f16(A, Bf0, acc0);
    acc1 = wmma_f16(A, Bf1, acc1);
  }
  float* Gp = gpart + ((size_t)part * B_ + b) * 4096;
#pragma unroll
  for (int r = 0; r < 8; ++r) {               // C/D layout: row = r + 8*hi
    int c = 16 * ci + r + 8 * hi;
    Gp[c * 64 + 16 * di0 + ln]      = acc0[r];
    Gp[c * 64 + 16 * di0 + 16 + ln] = acc1[r];
  }
}

// ===========================================================================
// K3a: hsum[b][c] = sum_t h'[b][c][t]
// ===========================================================================
__global__ void hsum_kernel(const _Float16* __restrict__ h16,
                            float* __restrict__ hsum)
{
  __shared__ float red[256];
  const int b = blockIdx.x, tid = threadIdx.x;
  const int c = tid & 63, seg = tid >> 6;
  const _Float16* row = h16 + ((size_t)b * 64 + c) * T_ + seg * (T_ / 4);
  float s = 0.f;
  for (int t = 0; t < T_ / 4; ++t) s += (float)row[t];
  red[tid] = s;
  __syncthreads();
  if (tid < 64)
    hsum[b * 64 + tid] = red[tid] + red[tid + 64] + red[tid + 128] + red[tid + 192];
}

// ===========================================================================
// K0: fold BN into C3/C4 (f16), compose collapsed final linear layer, and
// repack TCN block-3 weights into WMMA K-ordering (f16).
// ===========================================================================
__global__ void prep_kernel(
    const float* l1w, const float* l1b, const float* l2w, const float* l2b,
    const float* l3w, const float* l3b,
    const float* c3w, const float* c3b, const float* g3, const float* be3,
    const float* c4w, const float* c4b, const float* g4, const float* be4,
    const float* w1c, const float* w2c, const float* wdc,
    _Float16* C3f, _Float16* C4f, float* bias3, float* bias4,
    float* weff, float* beff,
    _Float16* w1cp, _Float16* w2cp, _Float16* wdcp)
{
  const int tid = threadIdx.x; // 256 threads
  if (tid < 64) {
    float acc = 0.f;
    for (int j = 0; j < 16; ++j) {
      float t2 = 0.f;
      for (int i = 0; i < 32; ++i) t2 += l2w[j * 32 + i] * l1w[i * 64 + tid];
      acc += l3w[j] * t2;
    }
    weff[tid] = acc;
  }
  if (tid == 0) {
    float acc = l3b[0];
    for (int j = 0; j < 16; ++j) {
      float t2 = l2b[j];
      for (int i = 0; i < 32; ++i) t2 += l2w[j * 32 + i] * l1b[i];
      acc += l3w[j] * t2;
    }
    *beff = acc;
  }
  for (int idx = tid; idx < 1024; idx += 256) {
    int j = idx >> 6;                       // C3: [16][64]
    C3f[idx] = (_Float16)(c3w[idx] * g3[j] * BN_SCALE_F);
    int c = idx >> 4;                       // C4: [64][16]
    C4f[idx] = (_Float16)(c4w[idx] * g4[c] * BN_SCALE_F);
  }
  if (tid < 16) bias3[tid] = c3b[tid] * g3[tid] * BN_SCALE_F + be3[tid];
  if (tid < 64) bias4[tid] = c4b[tid] * g4[tid] * BN_SCALE_F + be4[tid];
  // block-3 weight repack: K-major orderings matching the WMMA fragments
  for (int idx = tid; idx < 64 * 96; idx += 256) {        // K = k*32+ci
    int co = idx / 96, rem = idx % 96, k = rem >> 5, ci = rem & 31;
    w1cp[idx] = (_Float16)w1c[(co * 32 + ci) * 3 + k];
  }
  for (int idx = tid; idx < 64 * 192; idx += 256) {       // K = k*64+ci
    int co = idx / 192, rem = idx % 192, k = rem >> 6, ci = rem & 63;
    w2cp[idx] = (_Float16)w2c[(co * 64 + ci) * 3 + k];
  }
  for (int idx = tid; idx < 64 * 32; idx += 256)          // K = ci
    wdcp[idx] = (_Float16)wdc[idx];
}

// ===========================================================================
// K4: per-batch small math: scores from Gram, softmax, M1/M2 = S@Wv (f16),
// n1/n2 = S@bv, CAM avg-pool branch gate a.  64 threads (thread = row c).
// ===========================================================================
__global__ void attn_kernel(
    const float* __restrict__ gpart, const float* __restrict__ hsum,
    const float* Wq, const float* bq, const float* Wk, const float* bk,
    const float* Wv1, const float* bv1, const float* Wv2, const float* bv2,
    const float* c1w, const float* c1b, const float* g1, const float* be1,
    const float* c2w, const float* c2b, const float* g2, const float* be2,
    _Float16* __restrict__ M1f, _Float16* __restrict__ M2f,
    float* __restrict__ n1o, float* __restrict__ n2o, float* __restrict__ aGo)
{
  __shared__ float G[4096];
  __shared__ float tq[4096];
  __shared__ float sL[4096];
  __shared__ float hs[64], qh[64], kh[64], meanm[64], u[16];
  const int b = blockIdx.x, tid = threadIdx.x, c = tid;

  for (int idx = tid; idx < 4096; idx += 64) {
    float g = 0.f;
    for (int p = 0; p < 8; ++p) g += gpart[((size_t)p * B_ + b) * 4096 + idx];
    G[idx] = g;
  }
  hs[tid] = hsum[b * 64 + tid];
  __syncthreads();

  { float a = 0.f, k2 = 0.f;
    for (int e = 0; e < 64; ++e) { a += Wq[c * 64 + e] * hs[e]; k2 += Wk[c * 64 + e] * hs[e]; }
    qh[c] = a; kh[c] = k2; }
  __syncthreads();

  for (int f = 0; f < 64; ++f) {
    float a = 0.f;
    for (int e = 0; e < 64; ++e) a += Wq[c * 64 + e] * G[e * 64 + f];
    tq[c * 64 + f] = a;
  }
  float mx = -1e30f;
  for (int d = 0; d < 64; ++d) {
    float a = bq[c] * kh[d] + qh[c] * bk[d] + (float)T_ * bq[c] * bk[d];
    for (int f = 0; f < 64; ++f) a += tq[c * 64 + f] * Wk[d * 64 + f];
    sL[c * 64 + d] = a; mx = fmaxf(mx, a);
  }
  float den = 0.f;
  for (int d = 0; d < 64; ++d) { float e = __expf(sL[c * 64 + d] - mx); sL[c * 64 + d] = e; den += e; }
  float inv = 1.f / den;
  for (int d = 0; d < 64; ++d) sL[c * 64 + d] *= inv;

  float n1 = 0.f, n2 = 0.f, mm = 0.f;
  for (int d = 0; d < 64; ++d) { n1 += sL[c * 64 + d] * bv1[d]; n2 += sL[c * 64 + d] * bv2[d]; }
  for (int e = 0; e < 64; ++e) {
    float m1 = 0.f, m2 = 0.f;
    for (int d = 0; d < 64; ++d) {
      float s = sL[c * 64 + d];
      m1 += s * Wv1[d * 64 + e];
      m2 += s * Wv2[d * 64 + e];
    }
    M1f[(size_t)b * 4096 + c * 64 + e] = (_Float16)m1;
    M2f[(size_t)b * 4096 + c * 64 + e] = (_Float16)m2;
    mm += 0.5f * (m1 + m2) * (hs[e] * (1.f / (float)T_));
  }
  mm += 0.5f * (n1 + n2);
  n1o[b * 64 + c] = n1; n2o[b * 64 + c] = n2;
  meanm[c] = mm;
  __syncthreads();
  if (tid < 16) {
    float a = c1b[tid];
    for (int e = 0; e < 64; ++e) a += c1w[tid * 64 + e] * meanm[e];
    u[tid] = fmaxf(a * g1[tid] * BN_SCALE_F + be1[tid], 0.f);
  }
  __syncthreads();
  { float a = c2b[c];
    for (int j = 0; j < 16; ++j) a += c2w[c * 16 + j] * u[j];
    aGo[b * 64 + c] = a * g2[c] * BN_SCALE_F + be2[c]; }
}

// ===========================================================================
// K5: fused o1/o2 (WMMA) + CAM c-branch (WMMA C3, VALU C4) + gate + collapsed
// final MLP + sigmoid. One 16-timestep strip per wave, 8 waves/WG.
// ===========================================================================
__global__ void fused_out_kernel(
    const _Float16* __restrict__ h16T,
    const _Float16* __restrict__ M1f, const _Float16* __restrict__ M2f,
    const float* __restrict__ n1g, const float* __restrict__ n2g,
    const float* __restrict__ aGg,
    const _Float16* __restrict__ C3f, const _Float16* __restrict__ C4f,
    const float* __restrict__ bias3, const float* __restrict__ bias4,
    const float* __restrict__ weff, const float* __restrict__ beff,
    float* __restrict__ out)
{
  __shared__ __align__(32) _Float16 mT[8][16 * 64];   // m transposed [t][c]
  __shared__ __align__(32) _Float16 cbuf[8][16 * 16]; // cbr transposed [t][j]
  __shared__ float n1L[64], n2L[64], aGL[64], weffL[64], b4L[64], b3L[16];
  __shared__ float beffL;

  const int b = blockIdx.x;
  const int tid = threadIdx.x, lane = tid & 31, wave = tid >> 5;
  if (tid < 64) {
    n1L[tid] = n1g[b * 64 + tid];  n2L[tid] = n2g[b * 64 + tid];
    aGL[tid] = aGg[b * 64 + tid];  weffL[tid] = weff[tid];
    b4L[tid] = bias4[tid];
  }
  if (tid < 16) b3L[tid] = bias3[tid];
  if (tid == 0) beffL = *beff;
  __syncthreads();

  const int hi = (lane >> 4) & 1, ln = lane & 15;
  const int t0 = (blockIdx.y * 8 + wave) * 16;

  // B fragments of h' from [b][t][64] layout (contiguous 32B per lane)
  const _Float16* hb = h16T + ((size_t)b * T_ + t0) * 64;
  v16h Bh0 = *(const v16h*)(hb + ln * 64 + hi * 16);
  v16h Bh1 = *(const v16h*)(hb + ln * 64 + 32 + hi * 16);

  const _Float16* M1b = M1f + (size_t)b * 4096;
  const _Float16* M2b = M2f + (size_t)b * 4096;
  v8f o1[4], o2[4];
#pragma unroll
  for (int ci = 0; ci < 4; ++ci) {
    const _Float16* r1 = M1b + (16 * ci + ln) * 64;
    const _Float16* r2 = M2b + (16 * ci + ln) * 64;
    v8f a = {};
    a = wmma_f16(load_afrag(r1, 0, hi), Bh0, a);
    a = wmma_f16(load_afrag(r1, 32, hi), Bh1, a);
    o1[ci] = a;
    v8f a2 = {};
    a2 = wmma_f16(load_afrag(r2, 0, hi), Bh0, a2);
    a2 = wmma_f16(load_afrag(r2, 32, hi), Bh1, a2);
    o2[ci] = a2;
  }

  // biases + m -> LDS (transposed so C3's B-fragment reads are contiguous)
  _Float16* mw = mT[wave];
#pragma unroll
  for (int ci = 0; ci < 4; ++ci)
#pragma unroll
    for (int r = 0; r < 8; ++r) {
      int c = 16 * ci + r + 8 * hi;
      o1[ci][r] += n1L[c];
      o2[ci][r] += n2L[c];
      mw[ln * 64 + c] = (_Float16)(0.5f * (o1[ci][r] + o2[ci][r]));
    }

  // cbr = relu(C3' m + bias3) via WMMA (BN folded into C3')
  v8f cb = {};
  {
    const _Float16* r3 = C3f + ln * 64;
    v16h Bm0 = *(const v16h*)(mw + ln * 64 + hi * 16);
    v16h Bm1 = *(const v16h*)(mw + ln * 64 + 32 + hi * 16);
    cb = wmma_f16(load_afrag(r3, 0, hi), Bm0, cb);
    cb = wmma_f16(load_afrag(r3, 32, hi), Bm1, cb);
  }
  _Float16* cw = cbuf[wave];
#pragma unroll
  for (int r = 0; r < 8; ++r) {
    int j = r + 8 * hi;
    cw[ln * 16 + j] = (_Float16)fmaxf(cb[r] + b3L[j], 0.f);
  }

  // C4 tail + gating + collapsed final MLP (VALU on C/D-layout accumulators)
  union { v16h v; _Float16 e[16]; } myc;
  myc.v = *(const v16h*)(cw + ln * 16);
  float part = 0.f;
#pragma unroll
  for (int ci = 0; ci < 4; ++ci)
#pragma unroll
    for (int r = 0; r < 8; ++r) {
      int c = 16 * ci + r + 8 * hi;
      union { v16h v; _Float16 e[16]; } c4r;
      c4r.v = *(const v16h*)(C4f + c * 16);
      float cv = 0.f;
#pragma unroll
      for (int j = 0; j < 16; ++j) cv += (float)c4r.e[j] * (float)myc.e[j];
      cv += b4L[c];
      float g    = 1.f / (1.f + __expf(-(aGL[c] + cv)));
      float mo   = 0.5f * (o1[ci][r] + o2[ci][r]);
      float cam  = g * mo;                                   // _cam returns gate*x
      float taff = 0.5f * (cam * (o1[ci][r] - o2[ci][r]) + 1.f);
      part += weffL[c] * taff;
    }
  part += __shfl_xor(part, 16, 32);   // combine the two lane halves (same t)
  if (hi == 0)
    out[(size_t)b * T_ + t0 + ln] = 1.f / (1.f + __expf(-(part + beffL)));
}

// ===========================================================================
// Host launcher. Input order = jax pytree flatten of setup_inputs() dict:
// [x, point_label, label, params-leaves (alphabetical, nested), pe].
// ===========================================================================
extern "C" void kernel_launch(void* const* d_in, const int* in_sizes, int n_in,
                              void* d_out, int out_size, void* d_ws, size_t ws_size,
                              hipStream_t stream)
{
  (void)in_sizes; (void)n_in; (void)out_size; (void)ws_size;
  const float* x   = (const float*)d_in[0];
  // d_in[1]=point_label, d_in[2]=label : unused by the differentiable forward
  const float* bk  = (const float*)d_in[3];
  const float* bq  = (const float*)d_in[4];
  const float* bv1 = (const float*)d_in[5];
  const float* bv2 = (const float*)d_in[6];
  const float* be1 = (const float*)d_in[7];
  const float* be2 = (const float*)d_in[8];
  const float* be3 = (const float*)d_in[9];
  const float* be4 = (const float*)d_in[10];
  const float* c1b = (const float*)d_in[11];
  const float* c1w = (const float*)d_in[12];
  const float* c2b = (const float*)d_in[13];
  const float* c2w = (const float*)d_in[14];
  const float* c3b = (const float*)d_in[15];
  const float* c3w = (const float*)d_in[16];
  const float* c4b = (const float*)d_in[17];
  const float* c4w = (const float*)d_in[18];
  const float* g1  = (const float*)d_in[19];
  const float* g2  = (const float*)d_in[20];
  const float* g3  = (const float*)d_in[21];
  const float* g4  = (const float*)d_in[22];
  const float* l1b = (const float*)d_in[23];
  const float* l1w = (const float*)d_in[24];
  const float* l2b = (const float*)d_in[25];
  const float* l2w = (const float*)d_in[26];
  const float* l3b = (const float*)d_in[27];
  const float* l3w = (const float*)d_in[28];
  const float* t0b1 = (const float*)d_in[29];
  const float* t0b2 = (const float*)d_in[30];
  const float* t0bd = (const float*)d_in[31];
  const float* t0w1 = (const float*)d_in[32];
  const float* t0w2 = (const float*)d_in[33];
  const float* t0wd = (const float*)d_in[34];
  const float* t1b1 = (const float*)d_in[35];
  const float* t1b2 = (const float*)d_in[36];
  const float* t1bd = (const float*)d_in[37];
  const float* t1w1 = (const float*)d_in[38];
  const float* t1w2 = (const float*)d_in[39];
  const float* t1wd = (const float*)d_in[40];
  const float* t2b1 = (const float*)d_in[41];
  const float* t2b2 = (const float*)d_in[42];
  const float* t2bd = (const float*)d_in[43];
  const float* t2w1 = (const float*)d_in[44];
  const float* t2w2 = (const float*)d_in[45];
  const float* t2wd = (const float*)d_in[46];
  const float* wk  = (const float*)d_in[47];
  const float* wq  = (const float*)d_in[48];
  const float* wv1 = (const float*)d_in[49];
  const float* wv2 = (const float*)d_in[50];
  const float* pe  = (const float*)d_in[51];

  char* ws = (char*)d_ws;
  _Float16* h16   = (_Float16*)(ws + OFF_H16);
  _Float16* h16T  = (_Float16*)(ws + OFF_H16T);
  float*    gpart = (float*)(ws + OFF_GPART);
  float*    hsum  = (float*)(ws + OFF_HSUM);
  _Float16* M1f   = (_Float16*)(ws + OFF_M1);
  _Float16* M2f   = (_Float16*)(ws + OFF_M2);
  float*    n1    = (float*)(ws + OFF_N1);
  float*    n2    = (float*)(ws + OFF_N2);
  float*    aG    = (float*)(ws + OFF_AG);
  _Float16* C3f   = (_Float16*)(ws + OFF_C3F);
  _Float16* C4f   = (_Float16*)(ws + OFF_C4F);
  float*    b3    = (float*)(ws + OFF_B3);
  float*    b4    = (float*)(ws + OFF_B4);
  float*    weff  = (float*)(ws + OFF_WEFF);
  float*    beff  = (float*)(ws + OFF_BEFF);
  _Float16* w1cp  = (_Float16*)(ws + OFF_W1CP);
  _Float16* w2cp  = (_Float16*)(ws + OFF_W2CP);
  _Float16* wdcp  = (_Float16*)(ws + OFF_WDCP);
  float*    out   = (float*)d_out;

  prep_kernel<<<1, 256, 0, stream>>>(l1w, l1b, l2w, l2b, l3w, l3b,
                                     c3w, c3b, g3, be3, c4w, c4b, g4, be4,
                                     t2w1, t2w2, t2wd,
                                     C3f, C4f, b3, b4, weff, beff,
                                     w1cp, w2cp, wdcp);
  tcn_kernel<<<dim3(B_, T_ / 128), 256, 0, stream>>>(
      x, pe,
      t0w1, t0b1, t0w2, t0b2, t0wd, t0bd,
      t1w1, t1b1, t1w2, t1b2, t1wd, t1bd,
      t2b1, t2b2, t2bd,
      w1cp, w2cp, wdcp,
      h16, h16T);
  gram_kernel<<<dim3(B_, 8), 256, 0, stream>>>(h16, gpart);
  hsum_kernel<<<B_, 256, 0, stream>>>(h16, hsum);
  attn_kernel<<<B_, 64, 0, stream>>>(gpart, hsum,
                                     wq, bq, wk, bk, wv1, bv1, wv2, bv2,
                                     c1w, c1b, g1, be1, c2w, c2b, g2, be2,
                                     M1f, M2f, n1, n2, aG);
  fused_out_kernel<<<dim3(B_, T_ / 128), 256, 0, stream>>>(
      h16T, M1f, M2f, n1, n2, aG, C3f, C4f, b3, b4, weff, beff, out);
}
